// TR_embedding_80711025426882
// MI455X (gfx1250) — matrix operand order repaired and checked by reading, not compiled, not run on hardware
//
#include <hip/hip_runtime.h>
#include <hip/hip_bf16.h>

// Problem constants (from reference)
#define DL 128      // D
#define LL 256      // L
#define MSAC 256
#define PAIRC 256
#define TPLC 64
#define KMSA 44
#define NVOL 22
#define NT 4

typedef __attribute__((ext_vector_type(16))) _Float16 v16h;
typedef __attribute__((ext_vector_type(8)))  _Float16 v8h;
typedef __attribute__((ext_vector_type(8)))  float    v8f;

__device__ __forceinline__ int lane_id() { return (int)(threadIdx.x & 31u); }

// ---------------- CDNA5 async global->LDS copy (ASYNCcnt path) --------------
// Per-lane: LDS[vdst] = MEM[vaddr], 16 bytes/lane -> 512 B per wave-instruction.
__device__ __forceinline__ unsigned lds_off(const void* p) {
  return (unsigned)(uintptr_t)p;   // low 32 bits of generic LDS address = offset
}
__device__ __forceinline__ void async_b128(void* lds_dst, const void* gsrc) {
  asm volatile("global_load_async_to_lds_b128 %0, %1, off"
               :: "v"(lds_off(lds_dst)), "v"((unsigned long long)(uintptr_t)gsrc)
               : "memory");
}
__device__ __forceinline__ void wait_async0() {
  asm volatile("s_wait_asynccnt 0x0" ::: "memory");
}

// ---- WMMA fragment helpers (CDNA5 wave32 layouts, CDNA5 ISA) ----
// A 16x32 f16: lane&15 = row M, lane>>4 selects the K-octet inside each 16-K half.
// B 32x16 f16: lane&15 = col N, lane>>4 selects the K-octet inside each 16-K half.
// C/D v8f:     lanes 0-15 -> M 0-7, lanes 16-31 -> M 8-15; n = lane&15.

// Vectorized A loader from f32 source (global or LDS), fully in-bounds, 16B aligned.
__device__ __forceinline__ v16h load_a_f32_vec(const float* __restrict__ src,
                                               int stride, int k0) {
  const int ln  = lane_id();
  const float* p = src + (ln & 15) * stride + k0 + ((ln >> 4) << 3);
  const float4 x0 = *(const float4*)(p);        // h 0..3
  const float4 x1 = *(const float4*)(p + 4);    // h 4..7
  const float4 x2 = *(const float4*)(p + 16);   // h 8..11
  const float4 x3 = *(const float4*)(p + 20);   // h 12..15
  v16h a;
  a[0] = (_Float16)x0.x; a[1] = (_Float16)x0.y; a[2] = (_Float16)x0.z; a[3] = (_Float16)x0.w;
  a[4] = (_Float16)x1.x; a[5] = (_Float16)x1.y; a[6] = (_Float16)x1.z; a[7] = (_Float16)x1.w;
  a[8] = (_Float16)x2.x; a[9] = (_Float16)x2.y; a[10] = (_Float16)x2.z; a[11] = (_Float16)x2.w;
  a[12] = (_Float16)x3.x; a[13] = (_Float16)x3.y; a[14] = (_Float16)x3.z; a[15] = (_Float16)x3.w;
  return a;
}

// Guarded scalar A loader (for K tails that would read out of the allocation).
__device__ __forceinline__ v16h load_a_f32_guard(const float* __restrict__ src, int stride,
                                                 int k0, int kmax) {
  const int ln  = lane_id();
  const int row = ln & 15;
  const int kg  = (ln >> 4) << 3;
  v16h a;
#pragma unroll
  for (int h = 0; h < 16; ++h) {
    const int k = k0 + ((h < 8) ? (kg + h) : (16 + kg + (h - 8)));
    a[h] = (_Float16)((k < kmax) ? src[row * stride + k] : 0.0f);
  }
  return a;
}

// Vectorized A loader from f16 LDS source (16B aligned v8h chunks).
__device__ __forceinline__ v16h load_a_f16_vec(const _Float16* __restrict__ src,
                                               int stride, int k0) {
  const int ln = lane_id();
  const _Float16* p = src + (ln & 15) * stride + k0 + ((ln >> 4) << 3);
  const v8h lo = *(const v8h*)(p);        // h 0..7  (k = kg..kg+7)
  const v8h hi = *(const v8h*)(p + 16);   // h 8..15 (k = 16+kg..)
  v16h a;
#pragma unroll
  for (int h = 0; h < 8; ++h) { a[h] = lo[h]; a[h + 8] = hi[h]; }
  return a;
}

// Vectorized B loader: B[k][n] = W[n0+n][k], W row-major N x K (ld = K), in-bounds.
__device__ __forceinline__ v16h load_b_vec(const float* __restrict__ W, int ld,
                                           int n0, int k0) {
  const int ln = lane_id();
  const float* p = W + (n0 + (ln & 15)) * ld + k0 + ((ln >> 4) << 3);
  const float4 x0 = *(const float4*)(p);
  const float4 x1 = *(const float4*)(p + 4);
  const float4 x2 = *(const float4*)(p + 16);
  const float4 x3 = *(const float4*)(p + 20);
  v16h b;
  b[0] = (_Float16)x0.x; b[1] = (_Float16)x0.y; b[2] = (_Float16)x0.z; b[3] = (_Float16)x0.w;
  b[4] = (_Float16)x1.x; b[5] = (_Float16)x1.y; b[6] = (_Float16)x1.z; b[7] = (_Float16)x1.w;
  b[8] = (_Float16)x2.x; b[9] = (_Float16)x2.y; b[10] = (_Float16)x2.z; b[11] = (_Float16)x2.w;
  b[12] = (_Float16)x3.x; b[13] = (_Float16)x3.y; b[14] = (_Float16)x3.z; b[15] = (_Float16)x3.w;
  return b;
}

// Guarded scalar B loader (K tails).
__device__ __forceinline__ v16h load_b_guard(const float* __restrict__ W, int ld,
                                             int n0, int k0, int kmax) {
  const int ln = lane_id();
  const int n  = n0 + (ln & 15);
  const int kh = (ln >> 4) << 3;
  v16h b;
#pragma unroll
  for (int h = 0; h < 16; ++h) {
    const int k = k0 + ((h < 8) ? (kh + h) : (16 + kh + (h - 8)));
    b[h] = (_Float16)((k < kmax) ? W[n * ld + k] : 0.0f);
  }
  return b;
}

__device__ __forceinline__ v8f wmma16(v16h a, v16h b, v8f c) {
  return __builtin_amdgcn_wmma_f32_16x16x32_f16(false, a, false, b, (short)0, c, false, false);
}

// ---------------------------------------------------------------------------
// K1: seq projections (ql/qr/qm) + rq = LayerNorm(recycle_query)
__global__ __launch_bounds__(256) void k_precompute(
    const float* __restrict__ msa, const float* __restrict__ rquery,
    const float* __restrict__ W_ql, const float* __restrict__ b_ql,
    const float* __restrict__ W_qr, const float* __restrict__ b_qr,
    const float* __restrict__ W_qm, const float* __restrict__ b_qm,
    const float* __restrict__ g, const float* __restrict__ b,
    float* __restrict__ ql, float* __restrict__ qr,
    float* __restrict__ qm, float* __restrict__ rq) {
  const int i = blockIdx.x;
  const int c = threadIdx.x;
  __shared__ float seq[NVOL];
  __shared__ float red[256];
  if (c < NVOL) seq[c] = msa[i * (2 * NVOL) + c];   // msa_matrix[0, 0, i, c]

  const float x = rquery[i * MSAC + c];
  red[c] = x; __syncthreads();
  for (int s = 128; s > 0; s >>= 1) { if (c < s) red[c] += red[c + s]; __syncthreads(); }
  const float mean = red[0] * (1.0f / MSAC); __syncthreads();
  const float dm = x - mean;
  red[c] = dm * dm; __syncthreads();
  for (int s = 128; s > 0; s >>= 1) { if (c < s) red[c] += red[c + s]; __syncthreads(); }
  const float var = red[0] * (1.0f / MSAC);
  rq[i * MSAC + c] = dm * rsqrtf(var + 1e-5f) * g[c] + b[c];

  float aql = b_ql[c], aqr = b_qr[c], aqm = b_qm[c];
#pragma unroll
  for (int k = 0; k < NVOL; ++k) {
    const float s = seq[k];
    aql += s * W_ql[c * NVOL + k];
    aqr += s * W_qr[c * NVOL + k];
    aqm += s * W_qm[c * NVOL + k];
  }
  ql[i * PAIRC + c] = aql;
  qr[i * PAIRC + c] = aqr;
  qm[i * MSAC + c] = aqm;
}

// ---------------------------------------------------------------------------
// K2: msa = msa_matrix @ W_msa^T + b + qm[i] (+ rq[i] at d==0)
// grid (16, 512), block 128 (4 waves), one 16x16 tile per wave, K=44 (pad 64)
__global__ __launch_bounds__(128) void k_msa(
    const float* __restrict__ msa, const float* __restrict__ W_msa,
    const float* __restrict__ b_msa, const float* __restrict__ qm,
    const float* __restrict__ rq, float* __restrict__ out_msa) {
  const int wv = (int)(threadIdx.x >> 5);
  const int tile_m = blockIdx.y * 4 + wv;
  const int tile_n = blockIdx.x;
  const float* A = msa + (size_t)tile_m * 16 * KMSA;

  v8f c = {};
  {
    v16h a0 = load_a_f32_vec(A, KMSA, 0);                       // k 0..31 in-bounds
    v16h b0 = load_b_vec(W_msa, KMSA, tile_n * 16, 0);
    c = wmma16(a0, b0, c);
    v16h a1 = load_a_f32_guard(A, KMSA, 32, KMSA);              // k tail 32..43
    v16h b1 = load_b_guard(W_msa, KMSA, tile_n * 16, 32, KMSA);
    c = wmma16(a1, b1, c);
  }
  const int ln = lane_id();
  const int n  = tile_n * 16 + (ln & 15);
  const int mb = tile_m * 16 + ((ln >> 4) << 3);
#pragma unroll
  for (int r = 0; r < 8; ++r) {
    const int m = mb + r;
    const int d = m >> 8, i = m & 255;
    const float rqv = rq[i * MSAC + n];                         // always in-bounds
    float v = c[r] + b_msa[n] + qm[i * MSAC + n] + ((d == 0) ? rqv : 0.0f);
    out_msa[(size_t)m * MSAC + n] = v;
  }
}

// ---------------------------------------------------------------------------
// K3: rr base = ql[i] + qr[j] + PE[|i-j| clip 32] + LN(recycle_rr) + dist@W_dist^T + b
__global__ __launch_bounds__(256) void k_rr_base(
    const float* __restrict__ rr_in, const float* __restrict__ rdist,
    const float* __restrict__ ql, const float* __restrict__ qr,
    const float* __restrict__ PE, const float* __restrict__ g,
    const float* __restrict__ b, const float* __restrict__ W_dist,
    const float* __restrict__ b_dist, float* __restrict__ out_rr) {
  const int j = blockIdx.x, i = blockIdx.y, c = (int)threadIdx.x;
  __shared__ float red[256];
  __shared__ float dv[15];
  const size_t base = ((size_t)(i * LL + j)) * PAIRC;
  const float x = rr_in[base + c];
  if (c < 15) dv[c] = rdist[(size_t)(i * LL + j) * 15 + c];
  red[c] = x; __syncthreads();
  for (int s = 128; s > 0; s >>= 1) { if (c < s) red[c] += red[c + s]; __syncthreads(); }
  const float mean = red[0] * (1.0f / PAIRC); __syncthreads();
  const float dm = x - mean;
  red[c] = dm * dm; __syncthreads();
  for (int s = 128; s > 0; s >>= 1) { if (c < s) red[c] += red[c + s]; __syncthreads(); }
  const float var = red[0] * (1.0f / PAIRC);
  const float lnv = dm * rsqrtf(var + 1e-5f) * g[c] + b[c];
  float dd = b_dist[c];
#pragma unroll
  for (int k = 0; k < 15; ++k) dd += dv[k] * W_dist[c * 15 + k];
  int ss = i - j; if (ss < 0) ss = -ss; if (ss > 32) ss = 32;
  out_rr[base + c] = ql[i * PAIRC + c] + qr[j * PAIRC + c] + PE[ss * PAIRC + c] + lnv + dd;
}

// ---------------------------------------------------------------------------
// K4: fused pointwise attention, async-staged.
//  - rr-base tile (16x256 f32, 16KB) and template tiles (16x64 f32, 4KB) are
//    brought into LDS with GLOBAL_LOAD_ASYNC_TO_LDS_B128 (ASYNCcnt), template
//    tiles double-buffered so copy(t+1) overlaps WMMAs(t).
//  - q = rr@WQ^T (K split over waves, ds_add_f32 accumulate, scaled 1/4)
//  - per t: k,v = tpl@WK/WV^T; logits reduced from k fragments via shfl_xor;
//    v fragments stay in registers; softmax over t; x -> LDS f16;
//  - out_rr += x@WO^T + bO.
// grid 4096, block 256 (8 waves)
__global__ __launch_bounds__(256) void k_attn(
    const float* __restrict__ tpl, const float* __restrict__ WQ,
    const float* __restrict__ WK, const float* __restrict__ WV,
    const float* __restrict__ WO, const float* __restrict__ bO,
    float* __restrict__ out_rr) {
  const int tile = blockIdx.x;
  const int tid  = (int)threadIdx.x;
  const int wv   = tid >> 5;
  const int ln   = lane_id();
  const int i    = (tile * 16) >> 8;
  const int j0   = (tile * 16) & 255;
  const float* Abase = out_rr + (size_t)tile * 16 * PAIRC;

  __shared__ __align__(16) float a_rr[16 * PAIRC];      // 16 KB
  __shared__ __align__(16) float a_tp[2][16 * TPLC];    // 2 x 4 KB double buffer
  __shared__ float q_s[16 * TPLC];
  __shared__ float logit_s[NT][16][4];
  __shared__ float attn_s[NT][16][4];
  __shared__ __align__(16) _Float16 x_h[16 * TPLC];

  // ---- async stage: rr tile (1024 float4) + template t=0 (256 float4) ----
  {
#pragma unroll
    for (int u = 0; u < 4; ++u) {
      const int f4 = tid + u * 256;                     // float4 index
      async_b128(&a_rr[f4 * 4], Abase + f4 * 4);
    }
    const float* At0 = tpl + ((size_t)(0 * LL + i) * LL + j0) * TPLC;
    async_b128(&a_tp[0][tid * 4], At0 + tid * 4);
    for (int t = tid; t < 16 * TPLC; t += 256) q_s[t] = 0.0f;
    wait_async0();
  }
  __syncthreads();

  // ---- q projection from LDS: wave -> (group g = wv&3, K-half kh = wv>>2) ----
  {
    const int g = wv & 3, kh = wv >> 2;
    v8f c = {};
#pragma unroll
    for (int kk = 0; kk < 4; ++kk) {
      const int k0 = kh * 128 + kk * 32;
      v16h a = load_a_f32_vec(a_rr, PAIRC, k0);
      v16h b = load_b_vec(WQ, PAIRC, g * 16, k0);
      c = wmma16(a, b, c);
    }
    const int n  = g * 16 + (ln & 15);
    const int mb = (ln >> 4) << 3;
#pragma unroll
    for (int r = 0; r < 8; ++r)
      atomicAdd(&q_s[(mb + r) * TPLC + n], c[r] * 0.25f);  // 1/sqrt(PPG=16)
  }
  __syncthreads();

  // ---- k (waves 0-3) / v (waves 4-7) per template, double-buffered ----
  v8f vfrag[NT] = {};
#pragma unroll
  for (int t = 0; t < NT; ++t) {
    // prefetch next template tile while computing this one
    if (t + 1 < NT) {
      const float* Atn = tpl + ((size_t)((t + 1) * LL + i) * LL + j0) * TPLC;
      async_b128(&a_tp[(t + 1) & 1][tid * 4], Atn + tid * 4);
    }
    const int g = wv & 3;
    const float* At = a_tp[t & 1];
    const float* Wm = (wv < 4) ? WK : WV;
    v8f c = {};
    v16h a0 = load_a_f32_vec(At, TPLC, 0);
    v16h b0 = load_b_vec(Wm, TPLC, g * 16, 0);
    c = wmma16(a0, b0, c);
    v16h a1 = load_a_f32_vec(At, TPLC, 32);
    v16h b1 = load_b_vec(Wm, TPLC, g * 16, 32);
    c = wmma16(a1, b1, c);
    if (wv < 4) {
      const int n  = g * 16 + (ln & 15);
      const int mb = (ln >> 4) << 3;
#pragma unroll
      for (int r = 0; r < 8; ++r) {
        float p = q_s[(mb + r) * TPLC + n] * c[r];
        p += __shfl_xor(p, 1, 16);
        p += __shfl_xor(p, 2, 16);
        p += __shfl_xor(p, 4, 16);
        p += __shfl_xor(p, 8, 16);
        if ((ln & 15) == 0) logit_s[t][mb + r][g] = p;
      }
    } else {
      vfrag[t] = c;
    }
    if (t + 1 < NT) { wait_async0(); __syncthreads(); }
  }
  __syncthreads();

  // ---- softmax over t (4 entries) per (row m, group g) ----
  if (tid < 64) {
    const int m = tid >> 2, g = tid & 3;
    const float l0 = logit_s[0][m][g], l1 = logit_s[1][m][g];
    const float l2 = logit_s[2][m][g], l3 = logit_s[3][m][g];
    const float mx = fmaxf(fmaxf(l0, l1), fmaxf(l2, l3));
    const float e0 = __expf(l0 - mx), e1 = __expf(l1 - mx);
    const float e2 = __expf(l2 - mx), e3 = __expf(l3 - mx);
    const float inv = 1.0f / (e0 + e1 + e2 + e3);
    attn_s[0][m][g] = e0 * inv; attn_s[1][m][g] = e1 * inv;
    attn_s[2][m][g] = e2 * inv; attn_s[3][m][g] = e3 * inv;
  }
  __syncthreads();

  // ---- x = sum_t attn * v (waves 4-7, v fragments in registers) ----
  if (wv >= 4) {
    const int g  = wv & 3;
    const int n  = g * 16 + (ln & 15);
    const int mb = (ln >> 4) << 3;
#pragma unroll
    for (int r = 0; r < 8; ++r) {
      const int m = mb + r;
      const float xv = attn_s[0][m][g] * vfrag[0][r] + attn_s[1][m][g] * vfrag[1][r]
                     + attn_s[2][m][g] * vfrag[2][r] + attn_s[3][m][g] * vfrag[3][r];
      x_h[m * TPLC + n] = (_Float16)xv;
    }
  }
  __syncthreads();

  // ---- out projection: rr += x @ WO^T + bO ; each wave does 2 N-tiles ----
#pragma unroll
  for (int q2 = 0; q2 < 2; ++q2) {
    const int nt = wv * 2 + q2;
    v8f c = {};
    v16h a0 = load_a_f16_vec(x_h, TPLC, 0);
    v16h b0 = load_b_vec(WO, TPLC, nt * 16, 0);
    c = wmma16(a0, b0, c);
    v16h a1 = load_a_f16_vec(x_h, TPLC, 32);
    v16h b1 = load_b_vec(WO, TPLC, nt * 16, 32);
    c = wmma16(a1, b1, c);
    const int n  = nt * 16 + (ln & 15);
    const int mb = (ln >> 4) << 3;
#pragma unroll
    for (int r = 0; r < 8; ++r) {
      const size_t idx = (size_t)(tile * 16 + mb + r) * PAIRC + n;
      out_rr[idx] += c[r] + bO[n];
    }
  }
}

// ---------------------------------------------------------------------------
extern "C" void kernel_launch(void* const* d_in, const int* in_sizes, int n_in,
                              void* d_out, int out_size, void* d_ws, size_t ws_size,
                              hipStream_t stream) {
  (void)in_sizes; (void)n_in; (void)out_size; (void)ws_size;
  const float* msa     = (const float*)d_in[0];
  const float* tpl     = (const float*)d_in[1];
  const float* rquery  = (const float*)d_in[2];
  const float* rrin    = (const float*)d_in[3];
  const float* rdist   = (const float*)d_in[4];
  const float* W_msa   = (const float*)d_in[5];
  const float* b_msa   = (const float*)d_in[6];
  const float* W_ql    = (const float*)d_in[7];
  const float* b_ql    = (const float*)d_in[8];
  const float* W_qr    = (const float*)d_in[9];
  const float* b_qr    = (const float*)d_in[10];
  const float* W_qm    = (const float*)d_in[11];
  const float* b_qm    = (const float*)d_in[12];
  const float* PE      = (const float*)d_in[13];
  const float* ln_q_g  = (const float*)d_in[14];
  const float* ln_q_b  = (const float*)d_in[15];
  const float* ln_rr_g = (const float*)d_in[16];
  const float* ln_rr_b = (const float*)d_in[17];
  const float* W_dist  = (const float*)d_in[18];
  const float* b_dist  = (const float*)d_in[19];
  const float* WQ      = (const float*)d_in[20];
  const float* WK      = (const float*)d_in[21];
  const float* WV      = (const float*)d_in[22];
  const float* WO      = (const float*)d_in[23];
  const float* bO      = (const float*)d_in[24];

  float* out_msa = (float*)d_out;
  float* out_rr  = out_msa + (size_t)DL * LL * MSAC;

  float* ws = (float*)d_ws;
  float* ql = ws;
  float* qr = ws + 65536;
  float* qm = ws + 131072;
  float* rq = ws + 196608;

  k_precompute<<<256, 256, 0, stream>>>(msa, rquery, W_ql, b_ql, W_qr, b_qr,
                                        W_qm, b_qm, ln_q_g, ln_q_b, ql, qr, qm, rq);
  k_msa<<<dim3(16, 512), 128, 0, stream>>>(msa, W_msa, b_msa, qm, rq, out_msa);
  k_rr_base<<<dim3(256, 256), 256, 0, stream>>>(rrin, rdist, ql, qr, PE,
                                                ln_rr_g, ln_rr_b, W_dist, b_dist, out_rr);
  k_attn<<<4096, 256, 0, stream>>>(tpl, WQ, WK, WV, WO, bO, out_rr);
}
